// LaneAttentionGaussianPolicy_19610820674004
// MI455X (gfx1250) — matrix operand order repaired and checked by reading, not compiled, not run on hardware
//
#include <hip/hip_runtime.h>
#include <hip/hip_bf16.h>

// LaneAttentionGaussianPolicy for MI455X (gfx1250), wave32 + WMMA f16->f32.
// N=131072 TLs, L=4 lanes, E=64, H=128, NH=4, HD=16, NR=4, R=8.
// Each wave processes a group of 16 TLs. All GEMMs run as M=16 WMMA tiles.

#define NTL 131072

typedef __attribute__((ext_vector_type(16))) _Float16 v16h;
typedef __attribute__((ext_vector_type(8)))  float    v8f;

// d_ws layout (in f16 elements): pre-swizzled B fragments per GEMM.
#define OFF_W1    0        // lane_w1+b1 folded : Kpad=32,  N=128 -> 4096
#define OFF_W2    4096     // lane_w2           : K=128,    N=64  -> 8192
#define OFF_WIN   12288    // attn_in_w         : K=64,     N=192 -> 12288
#define OFF_WOUT  24576    // attn_out_w        : K=64,     N=64  -> 4096
#define OFF_WP1   28672    // phase_w1+b1 folded: Kpad=32,  N=128 -> 4096
#define OFF_WP2   32768    // phase_w2          : K=128,    N=128 -> 16384
#define OFF_WHEAD 49152    // head_w1 (Kpad=224): N=128           -> 28672
// total 77824 halves = 152KB of d_ws

// Map matrix element (row m, col c) -> index in fragment-major f16 buffer.
// Fragment convention (A operand, 16x16x32): lane l holds row m=l&15;
// halves j<8 -> K = 32*ks + (l>=16?8:0)+j ; halves j>=8 -> K = 32*ks + 16 + (l>=16?8:0)+(j-8)
__device__ __forceinline__ int frag_idx(int m, int c) {
  int ks = c >> 5, kk = c & 31, q = kk >> 3;
  int l = m + ((q & 1) << 4);
  int j = (kk & 7) + ((q >> 1) << 3);
  return ((ks << 5) + l) * 16 + j;
}

__device__ __forceinline__ void wave_sync() {
  __builtin_amdgcn_wave_barrier();
  asm volatile("s_wait_dscnt 0" ::: "memory");
  __builtin_amdgcn_wave_barrier();
}

__device__ __forceinline__ v8f wmma_f16(v16h a, v16h b, v8f c) {
  return __builtin_amdgcn_wmma_f32_16x16x32_f16(false, a, false, b, (short)0, c,
                                                false, false);
}

// GEMM with A fragments staged in LDS (frag-major), B fragments from global f16.
template<int KSTEPS, int NTILES, bool RELU>
__device__ __forceinline__ void gemm_lds(const _Float16* aBuf, const _Float16* bW,
                                         const float* bias, _Float16* outBuf,
                                         int outColOff, int lane) {
  const int n16 = lane & 15;
  const int mbase = (lane >> 4) << 3;
#pragma unroll
  for (int nt = 0; nt < NTILES; ++nt) {
    float bv = bias ? bias[nt * 16 + n16] : 0.f;
    v8f acc;
#pragma unroll
    for (int r = 0; r < 8; ++r) acc[r] = bv;
#pragma unroll
    for (int ks = 0; ks < KSTEPS; ++ks) {
      v16h a = *(const v16h*)(aBuf + (ks * 32 + lane) * 16);
      v16h b = *(const v16h*)(bW + ((ks * NTILES + nt) * 32 + lane) * 16);
      acc = wmma_f16(a, b, acc);
    }
    const int c = outColOff + nt * 16 + n16;
#pragma unroll
    for (int r = 0; r < 8; ++r) {
      float v = acc[r];
      if (RELU) v = fmaxf(v, 0.f);
      outBuf[frag_idx(mbase + r, c)] = (_Float16)v;
    }
  }
}

// Single-kstep GEMM with A fragment built directly in registers (K<=32).
template<int NTILES, bool RELU>
__device__ __forceinline__ void gemm_reg(v16h a, const _Float16* bW,
                                         _Float16* outBuf, int lane) {
  const int n16 = lane & 15;
  const int mbase = (lane >> 4) << 3;
#pragma unroll
  for (int nt = 0; nt < NTILES; ++nt) {
    v8f acc;
#pragma unroll
    for (int r = 0; r < 8; ++r) acc[r] = 0.f;
    v16h b = *(const v16h*)(bW + (nt * 32 + lane) * 16);
    acc = wmma_f16(a, b, acc);
    const int c = nt * 16 + n16;
#pragma unroll
    for (int r = 0; r < 8; ++r) {
      float v = acc[r];
      if (RELU) v = fmaxf(v, 0.f);
      outBuf[frag_idx(mbase + r, c)] = (_Float16)v;
    }
  }
}

__global__ __launch_bounds__(128) void lane_policy_kernel(
    const float* __restrict__ queue, const float* __restrict__ waiting,
    const float* __restrict__ phase_onehot, const float* __restrict__ elapsed,
    const int* __restrict__ region_ids, const float* __restrict__ noise,
    const float* __restrict__ lane_b2, const float* __restrict__ attn_in_b,
    const float* __restrict__ attn_out_b, const float* __restrict__ phase_b2,
    const float* __restrict__ region_table, const float* __restrict__ head_b1,
    const float* __restrict__ head_w2, const float* __restrict__ head_b2,
    const float* __restrict__ log_std, const _Float16* __restrict__ wb,
    float* __restrict__ out)
{
  __shared__ __align__(32) _Float16 smem[4 * 7680];  // 60KB, 4 waves/block
  const int lane = threadIdx.x & 31;
  const int warp = threadIdx.x >> 5;
  _Float16* buf0 = smem + warp * 7680;  // up to K=192 fragments (3072 halves)
  _Float16* buf1 = buf0 + 3072;         // up to K=64 fragments  (1024 halves)
  _Float16* feat = buf1 + 1024;         // K=224 head-input frags (3584 halves)
  const int base = (blockIdx.x * 4 + warp) * 16;  // first TL of this wave's group

  __builtin_prefetch(wb + OFF_W2, 0, 3);
  __builtin_prefetch(wb + OFF_WP2, 0, 3);
  __builtin_prefetch(wb + OFF_WHEAD, 0, 3);

  // ================= lane path: 4 iterations x 4 TLs (16 rows each) =========
  for (int it = 0; it < 4; ++it) {
    // L1 A fragment in registers: row = 4*tl + lane_id, K = [q, w, elapsed, 1, 0...]
    v16h a;
#pragma unroll
    for (int j = 0; j < 16; ++j) a[j] = (_Float16)0.f;
    if (lane < 16) {
      int t = lane >> 2, i = lane & 3;
      int n = base + it * 4 + t;
      a[0] = (_Float16)queue[n * 4 + i];
      a[1] = (_Float16)waiting[n * 4 + i];
      a[2] = (_Float16)elapsed[n];
      a[3] = (_Float16)1.f;             // bias row folded into W1
    }
    gemm_reg<8, true>(a, wb + OFF_W1, buf0, lane);                       // h  [16,128]
    wave_sync();
    gemm_lds<4, 4, true>(buf0, wb + OFF_W2, lane_b2, buf1, 0, lane);     // emb[16,64]
    wave_sync();
    gemm_lds<2, 12, false>(buf1, wb + OFF_WIN, attn_in_b, buf0, 0, lane);// qkv[16,192]
    wave_sync();

    // ---- attention (VALU): lane -> (tl, head, row-pair); tiny 4x4x16 ----
    {
      int t = lane >> 3, hd = (lane >> 1) & 3, i0 = (lane & 1) << 1;
      int r0 = t * 4 + i0, r1 = r0 + 1;
      float q0[16], q1[16];
#pragma unroll
      for (int d = 0; d < 16; ++d) {
        q0[d] = (float)buf0[frag_idx(r0, hd * 16 + d)];
        q1[d] = (float)buf0[frag_idx(r1, hd * 16 + d)];
      }
      float s0[4], s1[4];
#pragma unroll
      for (int j = 0; j < 4; ++j) {
        float a0 = 0.f, a1 = 0.f;
#pragma unroll
        for (int d = 0; d < 16; ++d) {
          float kv = (float)buf0[frag_idx(t * 4 + j, 64 + hd * 16 + d)];
          a0 += q0[d] * kv;
          a1 += q1[d] * kv;
        }
        s0[j] = a0 * 0.25f;             // 1/sqrt(HD)
        s1[j] = a1 * 0.25f;
      }
      float m0 = fmaxf(fmaxf(s0[0], s0[1]), fmaxf(s0[2], s0[3]));
      float m1 = fmaxf(fmaxf(s1[0], s1[1]), fmaxf(s1[2], s1[3]));
      float sum0 = 0.f, sum1 = 0.f;
#pragma unroll
      for (int j = 0; j < 4; ++j) {
        s0[j] = __expf(s0[j] - m0); sum0 += s0[j];
        s1[j] = __expf(s1[j] - m1); sum1 += s1[j];
      }
      float inv0 = 1.f / sum0, inv1 = 1.f / sum1;
      float o0[16], o1[16];
#pragma unroll
      for (int d = 0; d < 16; ++d) { o0[d] = 0.f; o1[d] = 0.f; }
#pragma unroll
      for (int j = 0; j < 4; ++j) {
        float w0 = s0[j] * inv0, w1 = s1[j] * inv1;
#pragma unroll
        for (int d = 0; d < 16; ++d) {
          float vv = (float)buf0[frag_idx(t * 4 + j, 128 + hd * 16 + d)];
          o0[d] += w0 * vv;
          o1[d] += w1 * vv;
        }
      }
#pragma unroll
      for (int d = 0; d < 16; ++d) {
        buf1[frag_idx(r0, hd * 16 + d)] = (_Float16)o0[d];
        buf1[frag_idx(r1, hd * 16 + d)] = (_Float16)o1[d];
      }
    }
    wave_sync();

    // ---- attn_out GEMM + mean over the 4 lanes -> ctx into feat cols [0,64) ----
    {
      const int n16 = lane & 15;
      const int tb = (lane >> 4) ? 2 : 0;  // low half owns TL 0,1; high half TL 2,3
#pragma unroll
      for (int nt = 0; nt < 4; ++nt) {
        float bv = attn_out_b[nt * 16 + n16];
        v8f acc;
#pragma unroll
        for (int r = 0; r < 8; ++r) acc[r] = bv;
#pragma unroll
        for (int ks = 0; ks < 2; ++ks) {
          v16h av = *(const v16h*)(buf1 + (ks * 32 + lane) * 16);
          v16h bm = *(const v16h*)(wb + OFF_WOUT + ((ks * 4 + nt) * 32 + lane) * 16);
          acc = wmma_f16(av, bm, acc);
        }
        int c = nt * 16 + n16;
        float c0 = (acc[0] + acc[1] + acc[2] + acc[3]) * 0.25f;
        float c1 = (acc[4] + acc[5] + acc[6] + acc[7]) * 0.25f;
        feat[frag_idx(it * 4 + tb, c)]     = (_Float16)c0;
        feat[frag_idx(it * 4 + tb + 1, c)] = (_Float16)c1;
      }
    }
    wave_sync();
  }

  // ================= phase path: all 16 TLs (M=16) ==========================
  {
    v16h a;
#pragma unroll
    for (int j = 0; j < 16; ++j) a[j] = (_Float16)0.f;
    if (lane < 16) {
      int n = base + lane;
#pragma unroll
      for (int j = 0; j < 4; ++j) a[j] = (_Float16)phase_onehot[n * 4 + j];
      a[4] = (_Float16)elapsed[n];
      a[5] = (_Float16)1.f;             // bias row folded into Wp1
    }
    gemm_reg<8, true>(a, wb + OFF_WP1, buf0, lane);                      // ph1 [16,128]
    wave_sync();
    gemm_lds<4, 8, true>(buf0, wb + OFF_WP2, phase_b2, feat, 64, lane);  // -> feat [64,192)
  }
  // region embedding -> feat cols [192,200); zero-pad [200,224)
  if (lane < 16) {
    int n = base + lane;
    int rid = region_ids[n];
    rid = rid < 0 ? 0 : (rid > 3 ? 3 : rid);
#pragma unroll
    for (int j = 0; j < 8; ++j)
      feat[frag_idx(lane, 192 + j)] = (_Float16)region_table[rid * 8 + j];
  }
  for (int x = lane; x < 24 * 16; x += 32)
    feat[frag_idx(x & 15, 200 + (x >> 4))] = (_Float16)0.f;
  wave_sync();

  // ================= head GEMM (K=224) fused with dot(head_w2) ==============
  float partial[8];
#pragma unroll
  for (int r = 0; r < 8; ++r) partial[r] = 0.f;
  {
    const int n16 = lane & 15;
#pragma unroll
    for (int nt = 0; nt < 8; ++nt) {
      float bvv = head_b1[nt * 16 + n16];
      v8f acc;
#pragma unroll
      for (int r = 0; r < 8; ++r) acc[r] = bvv;
#pragma unroll
      for (int ks = 0; ks < 7; ++ks) {
        v16h av = *(const v16h*)(feat + (ks * 32 + lane) * 16);
        v16h bm = *(const v16h*)(wb + OFF_WHEAD + ((ks * 8 + nt) * 32 + lane) * 16);
        acc = wmma_f16(av, bm, acc);
      }
      float w2c = head_w2[nt * 16 + n16];
#pragma unroll
      for (int r = 0; r < 8; ++r) partial[r] += fmaxf(acc[r], 0.f) * w2c;
    }
  }
  // reduce over the 16 lanes (columns) of each half-wave
#pragma unroll
  for (int r = 0; r < 8; ++r) {
    partial[r] += __shfl_xor(partial[r], 1);
    partial[r] += __shfl_xor(partial[r], 2);
    partial[r] += __shfl_xor(partial[r], 4);
    partial[r] += __shfl_xor(partial[r], 8);
  }

  float ls = log_std[0];
  float stdv = __expf(ls);
  if ((lane & 15) == 0) {               // lanes 0 and 16 write 8 TLs each
    int mb = (lane >> 4) << 3;
#pragma unroll
    for (int r = 0; r < 8; ++r) {
      int n = base + mb + r;
      float mean = partial[r] + head_b2[0];
      float nz = noise[n];
      float act = mean + stdv * nz;
      out[n] = fminf(fmaxf(act, -1.f), 1.f);
      // (actions - means) == std*noise exactly
      out[NTL + n] = -0.5f * (nz * nz + 2.f * ls + 1.8378770664093453f);
    }
  }
}

// Pre-swizzle one weight matrix into B-fragment order (f16), folding an
// optional bias row at K==biasK and zero-padding K up to Kpad.
__global__ void prep_weight_frags(const float* __restrict__ W,
                                  const float* __restrict__ biasRow,
                                  _Float16* __restrict__ dst,
                                  int Wrows, int N, int Kpad, int biasK) {
  int total = (Kpad >> 5) * (N >> 4) * 512;
  int t = blockIdx.x * blockDim.x + threadIdx.x;
  if (t >= total) return;
  int j = t & 15;
  int lane = (t >> 4) & 31;
  int tile = t >> 9;
  int ntiles = N >> 4;
  int nt = tile % ntiles;
  int ks = tile / ntiles;
  int n = nt * 16 + (lane & 15);
  int K = ks * 32 + ((lane >> 4) << 3) + (j & 7) + ((j >> 3) << 4);   // B operand: table order (8h + s, 16 + 8h + s-8)
  float v = 0.f;
  if (K < Wrows) v = W[K * N + n];
  else if (biasRow != nullptr && K == biasK) v = biasRow[n];
  dst[t] = (_Float16)v;
}

extern "C" void kernel_launch(void* const* d_in, const int* in_sizes, int n_in,
                              void* d_out, int out_size, void* d_ws, size_t ws_size,
                              hipStream_t stream) {
  (void)in_sizes; (void)n_in; (void)out_size; (void)ws_size;
  const float* queue        = (const float*)d_in[0];
  const float* waiting      = (const float*)d_in[1];
  const float* phase_onehot = (const float*)d_in[2];
  const float* elapsed      = (const float*)d_in[3];
  const int*   region_ids   = (const int*)d_in[4];
  const float* noise        = (const float*)d_in[5];
  const float* lane_w1      = (const float*)d_in[6];
  const float* lane_b1      = (const float*)d_in[7];
  const float* lane_w2      = (const float*)d_in[8];
  const float* lane_b2      = (const float*)d_in[9];
  const float* attn_in_w    = (const float*)d_in[10];
  const float* attn_in_b    = (const float*)d_in[11];
  const float* attn_out_w   = (const float*)d_in[12];
  const float* attn_out_b   = (const float*)d_in[13];
  const float* phase_w1     = (const float*)d_in[14];
  const float* phase_b1     = (const float*)d_in[15];
  const float* phase_w2     = (const float*)d_in[16];
  const float* phase_b2     = (const float*)d_in[17];
  const float* region_table = (const float*)d_in[18];
  const float* head_w1      = (const float*)d_in[19];
  const float* head_b1      = (const float*)d_in[20];
  const float* head_w2      = (const float*)d_in[21];
  const float* head_b2      = (const float*)d_in[22];
  const float* log_std      = (const float*)d_in[23];
  _Float16* wb = (_Float16*)d_ws;     // needs 77824 halves (152KB) of scratch
  float* out = (float*)d_out;

  auto prep = [&](const float* W, const float* b, int off,
                  int Wrows, int N, int Kpad, int biasK) {
    int total = (Kpad >> 5) * (N >> 4) * 512;
    int grid = (total + 255) / 256;
    prep_weight_frags<<<grid, 256, 0, stream>>>(W, b, wb + off, Wrows, N, Kpad, biasK);
  };
  prep(lane_w1,   lane_b1,  OFF_W1,    3,   128, 32,  3);
  prep(lane_w2,   nullptr,  OFF_W2,    128, 64,  128, -1);
  prep(attn_in_w, nullptr,  OFF_WIN,   64,  192, 64,  -1);
  prep(attn_out_w,nullptr,  OFF_WOUT,  64,  64,  64,  -1);
  prep(phase_w1,  phase_b1, OFF_WP1,   5,   128, 32,  5);
  prep(phase_w2,  nullptr,  OFF_WP2,   128, 128, 128, -1);
  prep(head_w1,   nullptr,  OFF_WHEAD, 200, 128, 224, -1);

  // 2048 blocks x 128 threads = 8192 waves; each wave handles 16 TLs.
  lane_policy_kernel<<<NTL / 64, 128, 0, stream>>>(
      queue, waiting, phase_onehot, elapsed, region_ids, noise,
      lane_b2, attn_in_b, attn_out_b, phase_b2, region_table,
      head_b1, head_w2, head_b2, log_std, wb, out);
}
